// emoplex_24180665876518
// MI455X (gfx1250) — matrix operand order repaired and checked by reading, not compiled, not run on hardware
//
#include <hip/hip_runtime.h>
#include <math.h>

#define T_ 110
#define B_ 256
#define D_ 512
#define H_ 512
#define BD_ (B_ * D_)           // 131072
#define TBD_ ((size_t)T_ * BD_) // 14417920
#define EPS_ 1e-9f

#define KSTEPS   (D_ / 32)      // 16 k-steps of 32
#define LDS_STRIDE 40           // 32 data halves + 8 pad -> 80B rows, conflict-free b128 reads
#define LDS_TILE (64 * LDS_STRIDE)

typedef __attribute__((ext_vector_type(16))) _Float16 v16h;
typedef __attribute__((ext_vector_type(8)))  _Float16 v8h;
typedef __attribute__((ext_vector_type(8)))  float    v8f;
typedef __attribute__((ext_vector_type(4)))  int      v4i;

typedef __attribute__((address_space(1))) v4i g_v4i;   // global <4 x i32>
typedef __attribute__((address_space(3))) v4i l_v4i;   // LDS    <4 x i32>

// ---------------------------------------------------------------------------
// Async global->LDS 16B copy (ASYNCcnt-tracked) + split-counter waits
// ---------------------------------------------------------------------------
__device__ __forceinline__ void async_copy16(const _Float16* g, _Float16* l)
{
#if __has_builtin(__builtin_amdgcn_global_load_async_to_lds_b128)
    __builtin_amdgcn_global_load_async_to_lds_b128(
        (g_v4i*)(uintptr_t)g,
        (l_v4i*)(uint32_t)(uintptr_t)l,   // flat addr[31:0] == LDS byte offset
        0, 0);
#else
    const uint32_t loff = (uint32_t)(uintptr_t)l;
    asm volatile("global_load_async_to_lds_b128 %0, %1, off"
                 :: "v"(loff), "v"((const void*)g) : "memory");
#endif
}

__device__ __forceinline__ void wait_async_le2()
{
#if __has_builtin(__builtin_amdgcn_s_wait_asynccnt)
    __builtin_amdgcn_s_wait_asynccnt(2);
#else
    asm volatile("s_wait_asynccnt 2" ::: "memory");
#endif
}

__device__ __forceinline__ void wait_async_le0()
{
#if __has_builtin(__builtin_amdgcn_s_wait_asynccnt)
    __builtin_amdgcn_s_wait_asynccnt(0);
#else
    asm volatile("s_wait_asynccnt 0" ::: "memory");
#endif
}

// ---------------------------------------------------------------------------
// Prep: fold W1/W2 into f16 Wp = W1+W2, Wm = W1-W2 (out = fr@Wp^T + fi@Wm^T + 2 b1)
// ---------------------------------------------------------------------------
__global__ void __launch_bounds__(256) prep_weights_kernel(
    const float* __restrict__ W1, const float* __restrict__ W2,
    _Float16* __restrict__ wp, _Float16* __restrict__ wm, int n)
{
    int i = blockIdx.x * blockDim.x + threadIdx.x;
    if (i < n) {
        float w1 = W1[i], w2 = W2[i];
        wp[i] = (_Float16)(w1 + w2);
        wm[i] = (_Float16)(w1 - w2);
    }
}

// ---------------------------------------------------------------------------
// Prep: cos/sin tables for turn rotations (per (t,d), shared across B)
// ---------------------------------------------------------------------------
__global__ void __launch_bounds__(256) prep_turn_kernel(
    const float* __restrict__ turna, const float* __restrict__ turnb,
    float* __restrict__ tac, float* __restrict__ tas,
    float* __restrict__ tbc, float* __restrict__ tbs, int n)
{
    int i = blockIdx.x * blockDim.x + threadIdx.x;
    if (i < n) {
        float sA, cA, sB, cB;
        __sincosf(turna[i], &sA, &cA);
        __sincosf(turnb[i], &sB, &cB);
        tac[i] = cA; tas[i] = sA;
        tbc[i] = cB; tbs[i] = sB;
    }
}

// ---------------------------------------------------------------------------
// Fused features + sequential scan. One thread per (b,d); pA/pB live in regs.
// Emits feats as f16 (real/imag planes) for the WMMA GEMM.
// ---------------------------------------------------------------------------
__global__ void __launch_bounds__(256) scan_kernel(
    const float* __restrict__ a, const float* __restrict__ v, const float* __restrict__ l,
    const float* __restrict__ qmask,
    const float* __restrict__ mu_a, const float* __restrict__ mu_v, const float* __restrict__ mu_l,
    const float* __restrict__ shifta, const float* __restrict__ shiftv,
    const float* __restrict__ tac, const float* __restrict__ tas,
    const float* __restrict__ tbc, const float* __restrict__ tbs,
    _Float16* __restrict__ frh, _Float16* __restrict__ fih)
{
    const int idx = blockIdx.x * blockDim.x + threadIdx.x; // b*D + d
    const int d = idx & (D_ - 1);
    const int b = idx >> 9;

    const float sa = shifta[d];
    const float sv = shiftv[d];

    float pAr = 0.f, pAi = 0.f, pBr = 0.f, pBi = 0.f;

    for (int t = 0; t < T_; ++t) {
        const size_t o = (size_t)t * BD_ + idx;
        const float ml = mu_l[t], ma = mu_a[t], mv = mu_v[t];

        float s0, c0, s1, c1, s2, c2;
        __sincosf(l[o],      &s0, &c0);
        __sincosf(a[o] + sa, &s1, &c1);
        __sincosf(v[o] + sv, &s2, &c2);
        const float fre = ml * c0 + ma * c1 + mv * c2;
        const float fim = ml * s0 + ma * s1 + mv * s2;

        const float qa = qmask[((size_t)t * B_ + b) * 2 + 0];
        const float qb = qmask[((size_t)t * B_ + b) * 2 + 1];

        float tr, ti;
        if (t == 0) {
            tr = fre; ti = fim;
            pAr = (1.f - qa) + qa * tr;  pAi = (1.f - qa) + qa * ti;
            pBr = (1.f - qb) + qb * tr;  pBi = (1.f - qb) + qb * ti;
        } else {
            const int to = t * D_ + d;
            const float ca = tac[to], sA = tas[to];
            const float cb = tbc[to], sB = tbs[to];
            // turn = qa * (pA * e^{i ta}) + qb * (pB * e^{i tb})
            const float rAr = pAr * ca - pAi * sA;
            const float rAi = pAr * sA + pAi * ca;
            const float rBr = pBr * cb - pBi * sB;
            const float rBi = pBr * sB + pBi * cb;
            float ur = qa * rAr + qb * rBr;
            float ui = qa * rAi + qb * rBi;
            const float m2 = ur * ur + ui * ui;
            if (m2 > 0.f) {
                const float inv = 1.0f / (sqrtf(m2) + EPS_);
                ur *= inv; ui *= inv;
            }
            tr = fre + ur; ti = fim + ui;
            pAr = (1.f - qa) * pAr + qa * tr;  pAi = (1.f - qa) * pAi + qa * ti;
            pBr = (1.f - qb) * pBr + qb * tr;  pBi = (1.f - qb) * pBi + qb * ti;
        }

        frh[o] = (_Float16)tr;
        fih[o] = (_Float16)ti;
    }
}

// ---------------------------------------------------------------------------
// WMMA GEMM with async double-buffered LDS A-tiles.
//   out[r, n] = sum_k fr[r,k]*Wp[n,k] + fi[r,k]*Wm[n,k] + 2*b1[n]
// Block = 8 waves: all share one 64-row A stripe (LDS), each wave owns one
// 16-col N tile -> block computes 64x128.  A-operand (16-bit, 16x32) per lane:
// row M=L%16; this code puts K=(L/16)*16+0..15 in elements 0-15 (same order as B).
// ---------------------------------------------------------------------------
__device__ __forceinline__ v16h load_a_pair(const _Float16* p)
{
    v8h lo = *(const v8h*)(p);       // K = kb .. kb+7
    v8h hi = *(const v8h*)(p + 8);   // K = kb+8 .. kb+15
    v16h r;
#pragma unroll
    for (int i = 0; i < 8; ++i) { r[i] = lo[i]; r[i + 8] = hi[i]; }
    return r;
}

__global__ void __launch_bounds__(256) gemm_kernel(
    const _Float16* __restrict__ frh, const _Float16* __restrict__ fih,
    const _Float16* __restrict__ wp,  const _Float16* __restrict__ wm,
    const float* __restrict__ b1, float* __restrict__ out)
{
    __shared__ __attribute__((aligned(32))) _Float16 ldsFr[2 * LDS_TILE];
    __shared__ __attribute__((aligned(32))) _Float16 ldsFi[2 * LDS_TILE];

    const int tid   = threadIdx.x;
    const int lane  = tid & 31;
    const int wave  = tid >> 5;
    const int mtile = blockIdx.x >> 2;               // 440 tiles of M=64
    const int ngrp  = blockIdx.x & 3;                // 4 groups of 8 N-tiles
    const int m0 = mtile * 64;
    const int n0 = (ngrp * 8 + wave) * 16;
    const int lmod  = lane & 15;
    const int lhalf = lane >> 4;

    // Fill geometry: 256 threads x 16B cover one 64x32 f16 plane per issue.
    const int frow = tid >> 2;          // 0..63
    const int fcol = (tid & 3) * 8;     // 0,8,16,24 (halves)
    const size_t gbase = (size_t)(m0 + frow) * D_ + fcol;
    const int    lbase = frow * LDS_STRIDE + fcol;

    // Prologue: fill both buffers (k-steps 0 and 1). 2 async ops per fill.
    async_copy16(frh + gbase,       &ldsFr[lbase]);
    async_copy16(fih + gbase,       &ldsFi[lbase]);
    async_copy16(frh + gbase + 32,  &ldsFr[LDS_TILE + lbase]);
    async_copy16(fih + gbase + 32,  &ldsFi[LDS_TILE + lbase]);

    const float bias = 2.0f * b1[n0 + lmod];
    v8f acc[4];
#pragma unroll
    for (int s = 0; s < 4; ++s)
#pragma unroll
        for (int i = 0; i < 8; ++i) acc[s][i] = bias;

    const size_t brow = (size_t)(n0 + lmod) * D_;    // weight row (row-major HxD)

#pragma unroll 1
    for (int kstep = 0; kstep < KSTEPS; ++kstep) {
        // Own fills for this buffer done; next buffer's 2 may stay in flight.
        if (kstep < KSTEPS - 1) wait_async_le2();
        else                    wait_async_le0();
        __syncthreads();                             // whole tile visible

        const int buf = kstep & 1;
        const _Float16* tFr = &ldsFr[buf * LDS_TILE];
        const _Float16* tFi = &ldsFi[buf * LDS_TILE];

        // B-operand (32x16): lane L holds col N=L%16, K=(L/16)*16+0..15 —
        // 16 contiguous halves of the Wp/Wm row (L2-resident).
        const int kbB = kstep * 32 + lhalf * 16;
        const v16h bP = *(const v16h*)(wp + brow + kbB);
        const v16h bM = *(const v16h*)(wm + brow + kbB);

        const int kbA = lhalf * 16;
#pragma unroll
        for (int s = 0; s < 4; ++s) {
            const int lrow = (s * 16 + lmod) * LDS_STRIDE + kbA;
            const v16h aR = load_a_pair(tFr + lrow);
            const v16h aI = load_a_pair(tFi + lrow);
            acc[s] = __builtin_amdgcn_wmma_f32_16x16x32_f16(
                false, aR, false, bP, (short)0, acc[s], false, false);
            acc[s] = __builtin_amdgcn_wmma_f32_16x16x32_f16(
                false, aI, false, bM, (short)0, acc[s], false, false);
        }

        __syncthreads();                             // all reads done before refill
        if (kstep + 2 < KSTEPS) {
            const size_t g = gbase + (size_t)(kstep + 2) * 32;
            async_copy16(frh + g, &ldsFr[buf * LDS_TILE + lbase]);
            async_copy16(fih + g, &ldsFi[buf * LDS_TILE + lbase]);
        }
    }

    // C/D layout: VGPR i, lanes 0-15 -> M = base+i, lanes 16-31 -> M = base+8+i; N = lane%16.
#pragma unroll
    for (int s = 0; s < 4; ++s) {
#pragma unroll
        for (int i = 0; i < 8; ++i) {
            const int row = m0 + s * 16 + lhalf * 8 + i;
            out[(size_t)row * H_ + n0 + lmod] = acc[s][i];
        }
    }
}

// ---------------------------------------------------------------------------
extern "C" void kernel_launch(void* const* d_in, const int* in_sizes, int n_in,
                              void* d_out, int out_size, void* d_ws, size_t ws_size,
                              hipStream_t stream) {
    const float* a      = (const float*)d_in[0];
    const float* v      = (const float*)d_in[1];
    const float* l      = (const float*)d_in[2];
    const float* qmask  = (const float*)d_in[3];
    // d_in[4] umask: unused in forward
    const float* mu_a   = (const float*)d_in[5];
    const float* mu_v   = (const float*)d_in[6];
    const float* mu_l   = (const float*)d_in[7];
    const float* shifta = (const float*)d_in[8];
    const float* shiftv = (const float*)d_in[9];
    const float* turna  = (const float*)d_in[10];
    const float* turnb  = (const float*)d_in[11];
    const float* W1     = (const float*)d_in[12];
    const float* b1     = (const float*)d_in[13];
    const float* W2     = (const float*)d_in[14];
    // d_in[15] b2: cancels in xr+xi; d_in[16] dia_len: fixed at T_
    float* out = (float*)d_out;

    // Workspace carve-out
    char* ws = (char*)d_ws;
    size_t off = 0;
    auto carve = [&](size_t bytes) {
        void* p = ws + off;
        off = (off + bytes + 255) & ~(size_t)255;
        return p;
    };
    _Float16* frh = (_Float16*)carve(TBD_ * sizeof(_Float16));
    _Float16* fih = (_Float16*)carve(TBD_ * sizeof(_Float16));
    _Float16* wp  = (_Float16*)carve((size_t)H_ * D_ * sizeof(_Float16));
    _Float16* wm  = (_Float16*)carve((size_t)H_ * D_ * sizeof(_Float16));
    float* tac = (float*)carve((size_t)T_ * D_ * sizeof(float));
    float* tas = (float*)carve((size_t)T_ * D_ * sizeof(float));
    float* tbc = (float*)carve((size_t)T_ * D_ * sizeof(float));
    float* tbs = (float*)carve((size_t)T_ * D_ * sizeof(float));

    const int nW = H_ * D_;                 // 262144
    prep_weights_kernel<<<(nW + 255) / 256, 256, 0, stream>>>(W1, W2, wp, wm, nW);

    const int nT = T_ * D_;                 // 56320
    prep_turn_kernel<<<(nT + 255) / 256, 256, 0, stream>>>(turna, turnb, tac, tas, tbc, tbs, nT);

    scan_kernel<<<BD_ / 256, 256, 0, stream>>>(a, v, l, qmask, mu_a, mu_v, mu_l,
                                               shifta, shiftv, tac, tas, tbc, tbs,
                                               frh, fih);

    // 440 M-supertiles(64) x 4 N-groups(8x16) = 1760 blocks of 8 waves
    gemm_kernel<<<440 * 4, 256, 0, stream>>>(frh, fih, wp, wm, b1, out);
}
